// SelfAttention_11175504904274
// MI455X (gfx1250) — matrix operand order repaired and checked by reading, not compiled, not run on hardware
//
#include <hip/hip_runtime.h>

// ---------------------------------------------------------------------------
// Multi-head self-attention for MI455X (gfx1250, wave32, WMMA).
// B=4, T=2048, E=1024, H=16, HD=64.
// Pipeline: [cvt Wo->f16] -> [q/k/v projections (WMMA)] -> [flash attention
// (WMMA + online softmax, async-to-LDS double-buffered K/V)] -> [output
// projection (WMMA)]. All matmuls via v_wmma_f32_16x16x32_f16 (f32 accum).
// Workspace layout (f16): qh 16MB | kh 16MB | vT 16MB | attn 16MB | Wo16 2MB.
// ---------------------------------------------------------------------------

#define BB 4
#define TT 2048
#define EE 1024
#define HH 16
#define HD 64

typedef __attribute__((ext_vector_type(16))) _Float16 v16h;
typedef __attribute__((ext_vector_type(8)))  _Float16 v8h;
typedef __attribute__((ext_vector_type(8)))  float    v8f;

__device__ __forceinline__ v8f wmma16(v16h a, v16h b, v8f c) {
    // D = A(16x32 f16) * B(32x16 f16) + C(16x16 f32)
    return __builtin_amdgcn_wmma_f32_16x16x32_f16(false, a, false, b,
                                                  (short)0, c, false, false);
}

// Reductions across the 16 lanes of a lane-half (XOR 1/2/4/8 leaves lane bit4
// untouched, so the two halves stay independent). Matches the WMMA C/D layout
// where one matrix row spans 16 lanes of one half.
__device__ __forceinline__ float half16_max(float v) {
    v = fmaxf(v, __int_as_float(__builtin_amdgcn_ds_swizzle(__float_as_int(v), 0x041F)));
    v = fmaxf(v, __int_as_float(__builtin_amdgcn_ds_swizzle(__float_as_int(v), 0x081F)));
    v = fmaxf(v, __int_as_float(__builtin_amdgcn_ds_swizzle(__float_as_int(v), 0x101F)));
    v = fmaxf(v, __int_as_float(__builtin_amdgcn_ds_swizzle(__float_as_int(v), 0x201F)));
    return v;
}
__device__ __forceinline__ float half16_sum(float v) {
    v += __int_as_float(__builtin_amdgcn_ds_swizzle(__float_as_int(v), 0x041F));
    v += __int_as_float(__builtin_amdgcn_ds_swizzle(__float_as_int(v), 0x081F));
    v += __int_as_float(__builtin_amdgcn_ds_swizzle(__float_as_int(v), 0x101F));
    v += __int_as_float(__builtin_amdgcn_ds_swizzle(__float_as_int(v), 0x201F));
    return v;
}

// Wait for all outstanding async (memory<->LDS) transfers of this wave.
__device__ __forceinline__ void wait_async0() {
    asm volatile("s_wait_asynccnt 0" ::: "memory");
}

// ---------------------------------------------------------------------------
// Kernel 0: fp32 -> f16 conversion (for Wo).
// ---------------------------------------------------------------------------
__global__ void cvt_f32_f16_kernel(const float* __restrict__ in,
                                   _Float16* __restrict__ out, int n) {
    for (int i = blockIdx.x * blockDim.x + threadIdx.x; i < n;
         i += gridDim.x * blockDim.x)
        out[i] = (_Float16)in[i];
}

// ---------------------------------------------------------------------------
// Kernel 1: per-head projection y[b,h,t,:] = xh[b,t,h,:] @ W^T  (shared W).
// GEMM of [B*T*H, 64] x [64, 64]. H==16 => one 16-row tile = the 16 heads of
// a single token (b,t); the whole A tile is one embedding row of x.
// transposed==0: out[b][h][t][d]   (row-major per head; for Q/K)
// transposed==1: out[b][h][d][t]   (V transposed for PV B-operand loads)
// ---------------------------------------------------------------------------
__global__ __launch_bounds__(256)
void proj_kernel(const float* __restrict__ x, const float* __restrict__ W,
                 _Float16* __restrict__ out, float scale, int transposed) {
    const int wave = threadIdx.x >> 5, lane = threadIdx.x & 31;
    const int col = lane & 15, hl = lane >> 4;
    const int mtile = blockIdx.x * 8 + wave;          // 8192 tiles total
    const int b = mtile / TT;
    const int t = mtile % TT;
    const float* xrow = x + ((size_t)b * TT + t) * EE + (size_t)col * HD;

    // A fragments (16x32 f16 layout): two contiguous 8-float chunks per half.
    v16h a0, a1;
    {
        _Float16* h0 = (_Float16*)&a0;
        _Float16* h1 = (_Float16*)&a1;
#pragma unroll
        for (int j = 0; j < 8; ++j) {
            h0[j]     = (_Float16)xrow[ 0 + hl * 8 + j];
            h0[j + 8] = (_Float16)xrow[16 + hl * 8 + j];
            h1[j]     = (_Float16)xrow[32 + hl * 8 + j];
            h1[j + 8] = (_Float16)xrow[48 + hl * 8 + j];
        }
    }

    v8f acc[4];
#pragma unroll
    for (int nt = 0; nt < 4; ++nt) {
        // B operand element j -> contraction e = k0 + 8*hl + j (j<8), k0 + 16 + 8*hl + (j-8) (j>=8), column = col.
        // B[e][d] = W[d][e] => read row d of W contiguously.
        v16h b0, b1;
        _Float16* bh0 = (_Float16*)&b0;
        _Float16* bh1 = (_Float16*)&b1;
        const float* wr = W + (size_t)(nt * 16 + col) * HD;
#pragma unroll
        for (int j = 0; j < 8; ++j) {
            bh0[j]     = (_Float16)wr[ 0 + 8 * hl + j];
            bh0[j + 8] = (_Float16)wr[16 + 8 * hl + j];
            bh1[j]     = (_Float16)wr[32 + 8 * hl + j];
            bh1[j + 8] = (_Float16)wr[48 + 8 * hl + j];
        }
        v8f c = {};
        c = wmma16(a0, b0, c);
        c = wmma16(a1, b1, c);
        acc[nt] = c;
    }

#pragma unroll
    for (int nt = 0; nt < 4; ++nt) {
#pragma unroll
        for (int e = 0; e < 8; ++e) {
            const int h = e + 8 * hl;      // row in tile == head index
            const int d = nt * 16 + col;
            const float v = acc[nt][e] * scale;
            size_t o;
            if (transposed) o = (((size_t)b * HH + h) * HD + d) * TT + t;
            else            o = (((size_t)b * HH + h) * TT + t) * HD + d;
            out[o] = (_Float16)v;
        }
    }
}

// ---------------------------------------------------------------------------
// Kernel 2: flash attention. Block = 4 waves; each wave owns a 16-row query
// tile (16x64 output). KV blocks of 64 keys: 8 WMMAs for S = Q*K^T, online
// softmax (once per 64 keys), 8 WMMAs for P*V. K/V tiles are streamed into
// double-buffered LDS with GLOBAL_LOAD_ASYNC_TO_LDS_B128 (ASYNCcnt) so the
// next tile's DMA overlaps the current tile's WMMAs.
// ---------------------------------------------------------------------------
typedef _Float16 TileRow[80];   // 160B rows: keeps v16h loads 32B-aligned

__device__ __forceinline__ void issue_tile_async(const _Float16* kp,
                                                 const _Float16* vp, int kv0,
                                                 TileRow* Kb, TileRow* Vb,
                                                 int tid) {
    const unsigned long long kbase = (unsigned long long)(uintptr_t)kp;
    const unsigned long long vbase = (unsigned long long)(uintptr_t)vp;
    // One 64x64-f16 tile = 512 16B chunks; 128 threads x 4 chunks, for each
    // of K and V^T. Each instruction copies 16B/lane straight into LDS.
#pragma unroll
    for (int c = 0; c < 4; ++c) {
        const int q = c * 128 + tid;
        const int r = q >> 3, cc = q & 7;          // row, 16B chunk in row
        const unsigned kl = (unsigned)(uintptr_t)&Kb[r][cc * 8];
        const unsigned ko = (unsigned)(((unsigned)(kv0 + r) * HD + cc * 8u) * 2u);
        asm volatile("global_load_async_to_lds_b128 %0, %1, %2"
                     :: "v"(kl), "v"(ko), "s"(kbase) : "memory");
        const unsigned vl = (unsigned)(uintptr_t)&Vb[r][cc * 8];
        const unsigned vo = (unsigned)(((unsigned)r * TT + (unsigned)kv0 + cc * 8u) * 2u);
        asm volatile("global_load_async_to_lds_b128 %0, %1, %2"
                     :: "v"(vl), "v"(vo), "s"(vbase) : "memory");
    }
}

__global__ __launch_bounds__(128)
void attn_kernel(const _Float16* __restrict__ q, const _Float16* __restrict__ k,
                 const _Float16* __restrict__ vt, _Float16* __restrict__ ao) {
    __shared__ __align__(32) _Float16 Kl[2][64][80];   // [buf][key][d]
    __shared__ __align__(32) _Float16 Vl[2][64][80];   // [buf][d][key]
    __shared__ __align__(32) _Float16 Pl[4][16][72];   // per-wave P staging

    const int tid = threadIdx.x;
    const int wave = tid >> 5, lane = tid & 31;
    const int col = lane & 15, hl = lane >> 4;
    const int qb = blockIdx.x & 31;                    // T/64 query blocks
    const int bh = blockIdx.x >> 5;                    // b*H + h
    const int b = bh >> 4, h = bh & 15;
    const int t0 = qb * 64 + wave * 16;

    const _Float16* qp = q  + (size_t)bh * TT * HD;
    const _Float16* kp = k  + (size_t)bh * TT * HD;
    const _Float16* vp = vt + (size_t)bh * HD * TT;    // row d, stride T

    // Kick off the DMA of the first K/V tile before touching Q.
    issue_tile_async(kp, vp, 0, Kl[0], Vl[0], tid);

    // Load this wave's Q tile into A-operand fragments (rows t0+col).
    v16h aq0, aq1;
    {
        const _Float16* qr = qp + (size_t)(t0 + col) * HD;
        *(v8h*)((_Float16*)&aq0)       = *(const v8h*)(qr +  0 + hl * 8);
        *(v8h*)(((_Float16*)&aq0) + 8) = *(const v8h*)(qr + 16 + hl * 8);
        *(v8h*)((_Float16*)&aq1)       = *(const v8h*)(qr + 32 + hl * 8);
        *(v8h*)(((_Float16*)&aq1) + 8) = *(const v8h*)(qr + 48 + hl * 8);
    }

    v8f o[4] = {};
    float rm[8], rs[8];
#pragma unroll
    for (int e = 0; e < 8; ++e) { rm[e] = -1e30f; rs[e] = 0.0f; }

    const int NIT = TT / 64;
    for (int kb = 0; kb < NIT; ++kb) {
        const int cur = kb & 1;
        // Tile kb has landed in LDS once our own asyncs retire; the barrier
        // publishes it to all waves AND guarantees everyone is done reading
        // the buffer we are about to overwrite with tile kb+1.
        wait_async0();
        __syncthreads();
        if (kb + 1 < NIT)
            issue_tile_async(kp, vp, (kb + 1) * 64, Kl[cur ^ 1], Vl[cur ^ 1], tid);

        TileRow* Kb = Kl[cur];
        TileRow* Vb = Vl[cur];

        // S = Q * K^T : four 16-key column tiles, head dim 64 = 2 WMMA chunks.
        v8f s[4];
#pragma unroll
        for (int nt = 0; nt < 4; ++nt) {
            v8f c = {};
            v16h bk;
            *(v8h*)((_Float16*)&bk) = *(const v8h*)&Kb[nt * 16 + col][8 * hl]; *(v8h*)(((_Float16*)&bk) + 8) = *(const v8h*)&Kb[nt * 16 + col][16 + 8 * hl]; c = wmma16(aq0, bk, c);
            *(v8h*)((_Float16*)&bk) = *(const v8h*)&Kb[nt * 16 + col][32 + 8 * hl]; *(v8h*)(((_Float16*)&bk) + 8) = *(const v8h*)&Kb[nt * 16 + col][48 + 8 * hl]; c = wmma16(aq1, bk, c);
            s[nt] = c;
        }

        // Online softmax over the 64 new keys (row = element e + 8*hl).
        float p[4][8], alpha[8];
#pragma unroll
        for (int e = 0; e < 8; ++e) {
            float me = fmaxf(fmaxf(s[0][e], s[1][e]), fmaxf(s[2][e], s[3][e]));
            me = half16_max(me);
            const float nm = fmaxf(rm[e], me);
            alpha[e] = __expf(rm[e] - nm);
            rm[e] = nm;
            float psum = 0.0f;
#pragma unroll
            for (int nt = 0; nt < 4; ++nt) {
                p[nt][e] = __expf(s[nt][e] - nm);
                psum += p[nt][e];
            }
            rs[e] = rs[e] * alpha[e] + half16_sum(psum);
        }
#pragma unroll
        for (int nt = 0; nt < 4; ++nt)
#pragma unroll
            for (int e = 0; e < 8; ++e) o[nt][e] *= alpha[e];

        // D-layout -> A-layout for P via per-wave LDS staging (in-order LDS).
#pragma unroll
        for (int nt = 0; nt < 4; ++nt)
#pragma unroll
            for (int e = 0; e < 8; ++e)
                Pl[wave][e + 8 * hl][nt * 16 + col] = (_Float16)p[nt][e];
        __builtin_amdgcn_wave_barrier();
        v16h ap0, ap1;
        *(v8h*)((_Float16*)&ap0)       = *(const v8h*)&Pl[wave][col][ 0 + hl * 8];
        *(v8h*)(((_Float16*)&ap0) + 8) = *(const v8h*)&Pl[wave][col][16 + hl * 8];
        *(v8h*)((_Float16*)&ap1)       = *(const v8h*)&Pl[wave][col][32 + hl * 8];
        *(v8h*)(((_Float16*)&ap1) + 8) = *(const v8h*)&Pl[wave][col][48 + hl * 8];

        // O += P * V : four 16-wide d tiles x two 32-key contraction chunks.
#pragma unroll
        for (int nt = 0; nt < 4; ++nt) {
            v16h bv;
            *(v8h*)((_Float16*)&bv) = *(const v8h*)&Vb[nt * 16 + col][8 * hl]; *(v8h*)(((_Float16*)&bv) + 8) = *(const v8h*)&Vb[nt * 16 + col][16 + 8 * hl]; o[nt] = wmma16(ap0, bv, o[nt]);
            *(v8h*)((_Float16*)&bv) = *(const v8h*)&Vb[nt * 16 + col][32 + 8 * hl]; *(v8h*)(((_Float16*)&bv) + 8) = *(const v8h*)&Vb[nt * 16 + col][48 + 8 * hl]; o[nt] = wmma16(ap1, bv, o[nt]);
        }
    }

    // Normalize and write heads recombined: ao[b][t][h*64 + d] (== [B,T,E]).
#pragma unroll
    for (int e = 0; e < 8; ++e) {
        const float inv = 1.0f / rs[e];
        const int t = t0 + e + 8 * hl;
        _Float16* orow = ao + ((size_t)b * TT + t) * EE + (size_t)h * HD;
#pragma unroll
        for (int nt = 0; nt < 4; ++nt)
            orow[nt * 16 + col] = (_Float16)(o[nt][e] * inv);
    }
}

// ---------------------------------------------------------------------------
// Kernel 3: output projection  out[M=B*T, E] = attn[M, E] @ Wo^T.
// B-operand fragments read directly from row-major f16 Wo (contiguous in k,
// 2MB => L2-resident and shared by all 8192 wave-tiles).
// ---------------------------------------------------------------------------
__global__ __launch_bounds__(256)
void oproj_kernel(const _Float16* __restrict__ a, const _Float16* __restrict__ w,
                  float* __restrict__ out) {
    const int wave = threadIdx.x >> 5, lane = threadIdx.x & 31;
    const int col = lane & 15, hl = lane >> 4;
    const int job = blockIdx.x * 8 + wave;   // 8192 = 512 mtiles * 16 ngroups
    const int m0 = (job >> 4) * 16;
    const int n0 = (job & 15) * 64;
    const _Float16* arow = a + (size_t)(m0 + col) * EE;

    v8f c0 = {}, c1 = {}, c2 = {}, c3 = {};
    for (int k0 = 0; k0 < EE; k0 += 32) {
        v16h af;
        *(v8h*)((_Float16*)&af)       = *(const v8h*)(arow + k0 + hl * 16);
        *(v8h*)(((_Float16*)&af) + 8) = *(const v8h*)(arow + k0 + 8 + hl * 16);
        const v16h b0 = *(const v16h*)(w + (size_t)(n0 +  0 + col) * EE + k0 + 16 * hl);
        const v16h b1 = *(const v16h*)(w + (size_t)(n0 + 16 + col) * EE + k0 + 16 * hl);
        const v16h b2 = *(const v16h*)(w + (size_t)(n0 + 32 + col) * EE + k0 + 16 * hl);
        const v16h b3 = *(const v16h*)(w + (size_t)(n0 + 48 + col) * EE + k0 + 16 * hl);
        c0 = wmma16(af, b0, c0);
        c1 = wmma16(af, b1, c1);
        c2 = wmma16(af, b2, c2);
        c3 = wmma16(af, b3, c3);
    }
#pragma unroll
    for (int e = 0; e < 8; ++e) {
        float* orow = out + (size_t)(m0 + e + 8 * hl) * EE + n0;
        orow[ 0 + col] = c0[e];
        orow[16 + col] = c1[e];
        orow[32 + col] = c2[e];
        orow[48 + col] = c3[e];
    }
}

// ---------------------------------------------------------------------------
extern "C" void kernel_launch(void* const* d_in, const int* in_sizes, int n_in,
                              void* d_out, int out_size, void* d_ws,
                              size_t ws_size, hipStream_t stream) {
    (void)in_sizes; (void)n_in; (void)out_size; (void)ws_size;
    const float* values = (const float*)d_in[0];
    const float* keys   = (const float*)d_in[1];
    const float* query  = (const float*)d_in[2];
    const float* Wv     = (const float*)d_in[3];
    const float* Wk     = (const float*)d_in[4];
    const float* Wq     = (const float*)d_in[5];
    const float* Wo     = (const float*)d_in[6];
    float* out = (float*)d_out;

    char* ws = (char*)d_ws;
    const size_t MB = 1024u * 1024u;
    _Float16* qh   = (_Float16*)(ws +  0 * MB);   // [B,H,T,HD] f16 (pre-scaled)
    _Float16* kh   = (_Float16*)(ws + 16 * MB);   // [B,H,T,HD] f16
    _Float16* vt   = (_Float16*)(ws + 32 * MB);   // [B,H,HD,T] f16 (transposed)
    _Float16* ah   = (_Float16*)(ws + 48 * MB);   // [B,T,E]    f16 attention out
    _Float16* wo16 = (_Float16*)(ws + 64 * MB);   // [E,E]      f16 Wo

    cvt_f32_f16_kernel<<<1024, 256, 0, stream>>>(Wo, wo16, EE * EE);
    proj_kernel<<<1024, 256, 0, stream>>>(query,  Wq, qh, 0.125f, 0); // 1/sqrt(64)
    proj_kernel<<<1024, 256, 0, stream>>>(keys,   Wk, kh, 1.0f,   0);
    proj_kernel<<<1024, 256, 0, stream>>>(values, Wv, vt, 1.0f,   1);
    attn_kernel<<<BB * HH * (TT / 64), 128, 0, stream>>>(qh, kh, vt, ah);
    oproj_kernel<<<1024, 256, 0, stream>>>(ah, wo16, out);
}
